// FlexibleDynamicsDecoderRNN_61091614818310
// MI455X (gfx1250) — matrix operand order repaired and checked by reading, not compiled, not run on hardware
//
#include <hip/hip_runtime.h>
#include <hip/hip_bf16.h>

// ---------------- problem constants (from reference) ----------------
#define BB   256          // batch
#define TT   1024         // timesteps
#define INP  32
#define ZZ   32
#define SS   32
#define HH   256          // hidden
#define OO   32           // output
#define DD   96           // INP+SS+ZZ, GRU input
#define G3   768          // 3*H

// f16 weight buffer offsets inside d_ws (element counts)
#define WIH_ELEMS  (G3*DD)        // 73728
#define WHH_ELEMS  (G3*HH)        // 196608
#define WOUT_ELEMS (OO*HH)        // 8192
#define WIH_OFF  0
#define WHH_OFF  (WIH_ELEMS)
#define WOUT_OFF (WIH_ELEMS + WHH_ELEMS)

// LDS strides (halves), padded vs. bank conflicts; multiples of 8 halves so
// every 16-byte fragment chunk stays 16B-aligned.
#define HSTR 264
#define XSTR 104

typedef __attribute__((ext_vector_type(16))) _Float16     v16h;
typedef __attribute__((ext_vector_type(8)))  float        v8f;
typedef __attribute__((ext_vector_type(4)))  unsigned int vu4;

// ---------------- fp32 -> fp16 weight conversion ----------------
__global__ void gru_convert_weights(const float* __restrict__ Wih,
                                    const float* __restrict__ Whh,
                                    const float* __restrict__ Wout,
                                    _Float16* __restrict__ wbuf) {
    int i = blockIdx.x * blockDim.x + threadIdx.x;
    int total = WIH_ELEMS + WHH_ELEMS + WOUT_ELEMS;
    if (i >= total) return;
    float v;
    if (i < WHH_OFF)            v = Wih[i - WIH_OFF];
    else if (i < WOUT_OFF)      v = Whh[i - WHH_OFF];
    else                        v = Wout[i - WOUT_OFF];
    wbuf[i] = (_Float16)v;
}

// ---------------- WMMA fragment helpers (wave32; contiguous k order, same in A and B) ----------------
// A-matrix 16x32 f16: lane holds row (lane&15); lanes<16 hold K {k0..+15},
// lanes>=16 hold K {k0+16..+31}. Two 16B chunks per lane.
__device__ __forceinline__ v16h load_a_lds(const _Float16* base, int row, int stride,
                                           int k0, int lane) {
    int hi = lane >> 4;
    const _Float16* p = base + row * stride + k0 + 16 * hi;
    union { v16h v; vu4 q[2]; } u;
    u.q[0] = *(const vu4*)(p);
    u.q[1] = *(const vu4*)(p + 8);
    return u.v;
}

// B-matrix 32x16 f16 from row-major W (x·W^T => B[k][n] = W[n][k]; W row n contiguous):
// lane holds column (lane&15) = W row jrow, K halves k0+16*(lane>>4) .. +16.
__device__ __forceinline__ v16h load_b_w(const _Float16* __restrict__ W, int ldk,
                                         int jrow, int k0, int lane) {
    const _Float16* p = W + (size_t)jrow * ldk + k0 + 16 * (lane >> 4);
    union { v16h v; vu4 q[2]; } u;
    u.q[0] = ((const vu4*)p)[0];
    u.q[1] = ((const vu4*)p)[1];
    return u.v;
}

// Volatile variant: forces a fresh L2 read each timestep so the compiler can
// neither hoist the loop-invariant weight loads out of the t-loop nor spill
// 190+ VGPRs of fragments to scratch.
__device__ __forceinline__ v16h load_b_wv(const _Float16* W, int ldk,
                                          int jrow, int k0, int lane) {
    const _Float16* p = W + (size_t)jrow * ldk + k0 + 16 * (lane >> 4);
    union { v16h v; vu4 q[2]; } u;
    u.q[0] = *(const volatile vu4*)(p);
    u.q[1] = *(const volatile vu4*)(p + 8);
    return u.v;
}

__device__ __forceinline__ v8f wmma16(v16h a, v16h b, v8f c) {
    return __builtin_amdgcn_wmma_f32_16x16x32_f16(false, a, false, b,
                                                  (short)0, c, false, false);
}

__device__ __forceinline__ float sigmoid_fast(float x) {
    return __builtin_amdgcn_rcpf(1.0f + __expf(-x));
}
__device__ __forceinline__ float tanh_fast(float x) {
    return 1.0f - 2.0f * __builtin_amdgcn_rcpf(__expf(2.0f * x) + 1.0f);
}

// ---------------- persistent GRU kernel ----------------
// grid = 16 blocks (one per 16-row batch tile), block = 512 threads (16 waves).
// Wave w owns hidden columns [16w,16w+16). Software-pipelined: h(t)·W_hh for
// step t+1 is computed in step t's tail, overlapped with the y-GEMM.
// The f32 master copy of h lives in registers (C-layout is lane-invariant
// across steps); only the f16 WMMA copy goes through LDS.
__global__ __launch_bounds__(512)
void gru_decoder_kernel(const float* __restrict__ init_y,
                        const float* __restrict__ obs,      // (B,T,INP)
                        const float* __restrict__ z_dyn,    // (B,Z)
                        const float* __restrict__ b_ih,     // (3H)
                        const float* __restrict__ b_hh,     // (3H)
                        const float* __restrict__ b_out,    // (O)
                        const _Float16* __restrict__ wbuf,  // f16 weights
                        float* __restrict__ out)            // (B,T,O)
{
    __shared__ __align__(16) _Float16 sh_h16[16 * HSTR];
    __shared__ __align__(16) _Float16 sh_x16[16 * XSTR];
    __shared__ __align__(16) float    sh_y  [16 * OO];
    __shared__ __align__(16) float    sh_z  [16 * ZZ];
    __shared__ float sh_bih[G3];
    __shared__ float sh_bhh[G3];
    __shared__ float sh_bo [OO];

    const int tid  = threadIdx.x;
    const int w    = tid >> 5;
    const int lane = tid & 31;
    const int li   = lane & 15;
    const int hi   = lane >> 4;
    const int b0   = blockIdx.x * 16;

    const _Float16* Wih  = wbuf + WIH_OFF;
    const _Float16* Whh  = wbuf + WHH_OFF;
    const _Float16* Wout = wbuf + WOUT_OFF;

    // ---- init LDS state ----
    for (int i = tid; i < G3; i += 512) { sh_bih[i] = b_ih[i]; sh_bhh[i] = b_hh[i]; }
    if (tid < OO) sh_bo[tid] = b_out[tid];
    for (int i = tid; i < 16 * 32; i += 512) {
        int m = i >> 5, c = i & 31;
        sh_y[i] = init_y[(size_t)(b0 + m) * SS + c];
        sh_z[i] = z_dyn [(size_t)(b0 + m) * ZZ + c];
    }
    for (int i = tid; i < 16 * HSTR; i += 512)
        sh_h16[i] = (_Float16)0.0f;

    const int jc = 16 * w + li;   // this wave's hidden column for lane li

    // W_ih fragments are small (9 frags = 72 VGPRs): register-resident for all steps.
    v16h wR[3], wZ[3], wN[3];
    #pragma unroll
    for (int kc = 0; kc < 3; ++kc) {
        wR[kc] = load_b_w(Wih, DD,          jc, kc * 32, lane);
        wZ[kc] = load_b_w(Wih, DD, HH     + jc, kc * 32, lane);
        wN[kc] = load_b_w(Wih, DD, 2 * HH + jc, kc * 32, lane);
    }
    __syncthreads();

    const float biR = sh_bih[jc],        bhR = sh_bhh[jc];
    const float biZ = sh_bih[HH  + jc],  bhZ = sh_bhh[HH  + jc];
    const float biN = sh_bih[2*HH + jc], bhN = sh_bhh[2*HH + jc];

    // Pipelined h-contribution accumulators: h(t-1)·W_hh per gate. h(-1)=0.
    v8f hAccR = {}, hAccZ = {}, hAccN = {};
    // f32 master copy of h for this lane's (row v+8*hi, col jc) elements.
    v8f hPrev = {};

    for (int t = 0; t < TT; ++t) {
        // ---- phase 1: build x(t) = [obs_t | y(t-1) | z_dyn] as f16 ----
        // wave w owns row w; lane owns columns {lane, lane+32, lane+64}.
        {
            const float* orow = &obs[((size_t)(b0 + w) * TT + t) * INP];
            sh_x16[w * XSTR + lane]      = (_Float16)orow[lane];          // coalesced
            sh_x16[w * XSTR + 32 + lane] = (_Float16)sh_y[w * OO + lane];
            sh_x16[w * XSTR + 64 + lane] = (_Float16)sh_z[w * ZZ + lane];
            if (lane == 0 && t + 1 < TT)   // global_prefetch_b8 of next obs row
                __builtin_prefetch(&obs[((size_t)(b0 + w) * TT + (t + 1)) * INP], 0, 3);
        }
        __syncthreads();

        // ---- phase 2: x-GEMM (9 WMMAs) on top of pipelined h-contribution ----
        v8f accR = hAccR, accZ = hAccZ, accXN = {};
        #pragma unroll
        for (int kc = 0; kc < 3; ++kc) {
            v16h a = load_a_lds(sh_x16, li, XSTR, kc * 32, lane);
            accR  = wmma16(a, wR[kc], accR);
            accZ  = wmma16(a, wZ[kc], accZ);
            accXN = wmma16(a, wN[kc], accXN);
        }

        // gates + hidden update (C layout: vgpr v -> row v+8*hi, col = li);
        // h_old comes straight from registers (hPrev), no LDS round-trip.
        #pragma unroll
        for (int v = 0; v < 8; ++v) {
            int m = v + 8 * hi;
            float r  = sigmoid_fast(accR[v] + biR + bhR);
            float zg = sigmoid_fast(accZ[v] + biZ + bhZ);
            float n  = tanh_fast(accXN[v] + biN + r * (hAccN[v] + bhN));
            float hn = (1.0f - zg) * n + zg * hPrev[v];
            hPrev[v] = hn;
            sh_h16[m * HSTR + jc] = (_Float16)hn;
        }
        __syncthreads();   // h(t) complete across all column groups

        // ---- phase 3: y-GEMM (waves 0,1) overlapped with h(t)·W_hh for t+1 ----
        if (w < 2) {
            v8f acc = {};
            #pragma unroll
            for (int kc = 0; kc < 8; ++kc) {
                v16h a = load_a_lds(sh_h16, li, HSTR, kc * 32, lane);
                v16h b = load_b_wv(Wout, HH, 16 * w + li, kc * 32, lane);
                acc = wmma16(a, b, acc);
            }
            int col = 16 * w + li;
            float bo = sh_bo[col];
            #pragma unroll
            for (int v = 0; v < 8; ++v) {
                int m = v + 8 * hi;
                float yv = acc[v] + bo;
                sh_y[m * OO + col] = yv;
                out[((size_t)(b0 + m) * TT + t) * OO + col] = yv;
            }
        }
        if (t + 1 < TT) {
            v8f nR = {}, nZ = {}, nN = {};
            #pragma unroll
            for (int kc = 0; kc < 8; ++kc) {
                int k0 = kc * 32;
                v16h a  = load_a_lds(sh_h16, li, HSTR, k0, lane);
                v16h bR = load_b_wv(Whh, HH,          jc, k0, lane);
                nR = wmma16(a, bR, nR);
                v16h bZ = load_b_wv(Whh, HH, HH     + jc, k0, lane);
                nZ = wmma16(a, bZ, nZ);
                v16h bN = load_b_wv(Whh, HH, 2 * HH + jc, k0, lane);
                nN = wmma16(a, bN, nN);
            }
            hAccR = nR; hAccZ = nZ; hAccN = nN;
        }
        __syncthreads();   // y(t) visible for next step's x build
    }
}

// ---------------- host launcher ----------------
extern "C" void kernel_launch(void* const* d_in, const int* in_sizes, int n_in,
                              void* d_out, int out_size, void* d_ws, size_t ws_size,
                              hipStream_t stream) {
    const float* init_y = (const float*)d_in[0];
    const float* obs    = (const float*)d_in[1];
    const float* z_dyn  = (const float*)d_in[2];
    const float* W_ih   = (const float*)d_in[3];
    const float* W_hh   = (const float*)d_in[4];
    const float* b_ih   = (const float*)d_in[5];
    const float* b_hh   = (const float*)d_in[6];
    const float* W_out  = (const float*)d_in[7];
    const float* b_out  = (const float*)d_in[8];
    float*       out    = (float*)d_out;
    _Float16*    wbuf   = (_Float16*)d_ws;   // needs ~557 KB

    int total = WIH_ELEMS + WHH_ELEMS + WOUT_ELEMS;
    gru_convert_weights<<<(total + 255) / 256, 256, 0, stream>>>(W_ih, W_hh, W_out, wbuf);

    gru_decoder_kernel<<<BB / 16, 512, 0, stream>>>(init_y, obs, z_dyn,
                                                    b_ih, b_hh, b_out, wbuf, out);
}
